// SwinMambaBlock_28157805592952
// MI455X (gfx1250) — compile-verified, matrix operand order checked by reading, not run on hardware
//
#include <hip/hip_runtime.h>
#include <hip/hip_bf16.h>
#include <math.h>

// ---------------------------------------------------------------------------
// SwinMamba block for MI455X (gfx1250, wave32, WMMA).
// All GEMMs -> v_wmma_f32_16x16x32_f16, 1x4 (1x3) tile strip per wave,
// compile-time K (template) with explicit one-chunk-ahead double buffering so
// loads stay in flight across each WMMA burst (incremental s_wait_loadcnt).
// Sequential SSM scan -> VALU kernel, state in registers, B/C via LDS.
// ---------------------------------------------------------------------------

typedef __attribute__((ext_vector_type(16))) _Float16 v16h;
typedef __attribute__((ext_vector_type(8)))  float    v8f;

#define DIMC     192
#define D_INNER  384
#define D_STATE  16
#define NTOK     100352L          // 32 * 56 * 56
#define NWIN     2048             // 32 * 8 * 8
#define WLEN     49
#define MTILES   6272             // NTOK / 16

// ---------------------------------------------------------------------------
// Weight pack: fp32 W[K x N] -> f16 Wt[Npad x Kpad] (transposed, zero padded)
// so the WMMA B operand is a contiguous 16-half load per lane.
// ---------------------------------------------------------------------------
__global__ void pack_weight(const float* __restrict__ W, _Float16* __restrict__ Wt,
                            int K, int N, int Kpad, int Npad) {
    int idx = blockIdx.x * 256 + threadIdx.x;
    if (idx >= Kpad * Npad) return;
    int n = idx / Kpad, k = idx % Kpad;
    float v = (k < K && n < N) ? W[(long)k * N + n] : 0.0f;
    Wt[idx] = (_Float16)v;
}

// ---------------------------------------------------------------------------
// Block reduction over 192 threads (6 waves of wave32)
// ---------------------------------------------------------------------------
__device__ inline float blockSum192(float v, float* lds) {
    #pragma unroll
    for (int o = 16; o > 0; o >>= 1) v += __shfl_xor(v, o, 32);
    int wid = threadIdx.x >> 5;
    if ((threadIdx.x & 31) == 0) lds[wid] = v;
    __syncthreads();
    float t = 0.f;
    #pragma unroll
    for (int i = 0; i < 6; ++i) t += lds[i];
    __syncthreads();
    return t;
}

// LN1 + window partition + LN2, write f16 activations in window order.
__global__ __launch_bounds__(192)
void ln_window_kernel(const float* __restrict__ x,
                      const float* __restrict__ g1, const float* __restrict__ b1,
                      const float* __restrict__ gm, const float* __restrict__ bm,
                      _Float16* __restrict__ hs16) {
    __shared__ float lds[8];
    const long tok = blockIdx.x;
    const int  c   = threadIdx.x;
    float v  = x[tok * DIMC + c];
    float mu = blockSum192(v, lds) * (1.0f / 192.0f);
    float d  = v - mu;
    float va = blockSum192(d * d, lds) * (1.0f / 192.0f);
    float h  = d * rsqrtf(va + 1e-5f) * g1[c] + b1[c];
    float mu2 = blockSum192(h, lds) * (1.0f / 192.0f);
    float d2  = h - mu2;
    float va2 = blockSum192(d2 * d2, lds) * (1.0f / 192.0f);
    float hs  = d2 * rsqrtf(va2 + 1e-5f) * gm[c] + bm[c];
    int b = (int)(tok / 3136), p = (int)(tok % 3136);
    int yy = p / 56, xx = p % 56;
    long w = (long)b * 64 + (yy / 7) * 8 + (xx / 7);
    int  t = (yy % 7) * 7 + (xx % 7);
    hs16[(w * WLEN + t) * DIMC + c] = (_Float16)hs;
}

// LN (natural token order) -> f16
__global__ __launch_bounds__(192)
void ln_kernel(const float* __restrict__ x,
               const float* __restrict__ g, const float* __restrict__ b,
               _Float16* __restrict__ out16) {
    __shared__ float lds[8];
    const long tok = blockIdx.x;
    const int  c   = threadIdx.x;
    float v  = x[tok * DIMC + c];
    float mu = blockSum192(v, lds) * (1.0f / 192.0f);
    float d  = v - mu;
    float va = blockSum192(d * d, lds) * (1.0f / 192.0f);
    out16[tok * DIMC + c] = (_Float16)(d * rsqrtf(va + 1e-5f) * g[c] + b[c]);
}

// ---------------------------------------------------------------------------
// Generic WMMA GEMM: C[M x N] = A[M x KP](f16) * Bt[N x KP]^T(f16)
// One 16x(16*NT) tile strip per wave; KP is compile-time so the chunk loop
// fully unrolls; chunk c+1's loads are issued before chunk c's WMMA burst
// (software pipeline -> partial s_wait_loadcnt). Epilogue fused via MODE:
//   MODE 0: W_in     -> u (fp32, n<384) | silu(z) (f16, n>=384)
//   MODE 1: W_xp     -> proj fp32 (ld 48) + dt-A operand f16 (ld 32, cols<12)
//   MODE 2: W_dt     -> dt_raw fp32 = v + b_dt[n]
//   MODE 3: W_out    -> x2 = x + window_reverse(v)   (residual scatter)
//   MODE 4: fc1      -> act16 = gelu(v + b)          (f16)
//   MODE 5: fc2      -> out = x2 + v + b             (fp32, final output)
// ---------------------------------------------------------------------------
template <int MODE, int NT, int KP>
__global__ __launch_bounds__(256)
void wmma_gemm(const _Float16* __restrict__ A, const _Float16* __restrict__ Bt,
               int NSuper,
               const float* __restrict__ f0, const float* __restrict__ f1,
               float* __restrict__ o0, _Float16* __restrict__ o1) {
    constexpr int NCHUNK = KP / 32;
    const int wave = threadIdx.x >> 5;
    const int lane = threadIdx.x & 31;
    const long tile = (long)blockIdx.x * 8 + wave;
    const int nts = (int)(tile % NSuper);        // N super-tile (NT tiles wide)
    const int mt  = (int)(tile / NSuper);
    const int lr  = lane & 15;                   // row in A tile / col in B tile
    const int ko  = (lane >> 4) << 4;            // K sub-block 0 or 16

    const _Float16* ap = A  + (long)(mt * 16 + lr) * KP + ko;
    const _Float16* bp = Bt + (long)(nts * NT * 16 + lr) * KP + ko;

    v8f acc[NT];
    #pragma unroll
    for (int j = 0; j < NT; ++j) acc[j] = (v8f){};

    // software pipeline: chunk c+1 loads issued before chunk c's WMMA burst
    v16h aCur, aNxt;
    v16h bCur[NT], bNxt[NT];
    aCur = *(const v16h*)(ap);
    #pragma unroll
    for (int j = 0; j < NT; ++j)
        bCur[j] = *(const v16h*)(bp + (long)j * 16 * KP);

    #pragma unroll
    for (int c = 0; c < NCHUNK; ++c) {
        if (c + 1 < NCHUNK) {
            const int k = (c + 1) * 32;
            __builtin_prefetch(ap + k + 224, 0, 1);     // global_prefetch_b8
            aNxt = *(const v16h*)(ap + k);
            #pragma unroll
            for (int j = 0; j < NT; ++j)
                bNxt[j] = *(const v16h*)(bp + (long)j * 16 * KP + k);
        }
        #pragma unroll
        for (int j = 0; j < NT; ++j)
            acc[j] = __builtin_amdgcn_wmma_f32_16x16x32_f16(
                false, aCur, false, bCur[j], (short)0, acc[j], false, false);
        if (c + 1 < NCHUNK) {
            aCur = aNxt;
            #pragma unroll
            for (int j = 0; j < NT; ++j) bCur[j] = bNxt[j];
        }
    }

    const int mbase = mt * 16 + ((lane >> 4) << 3);
    #pragma unroll
    for (int j = 0; j < NT; ++j) {
        const int n = (nts * NT + j) * 16 + lr;
        #pragma unroll
        for (int i = 0; i < 8; ++i) {
            const long m = mbase + i;
            float v = acc[j][i];
            if (MODE == 0) {
                if (n < D_INNER) o0[m * D_INNER + n] = v;
                else {
                    float s = v / (1.0f + __expf(-v));          // silu(z)
                    o1[m * D_INNER + (n - D_INNER)] = (_Float16)s;
                }
            } else if (MODE == 1) {
                o0[m * 48 + n] = v;                              // proj (ld 48)
                if (n < 32) o1[m * 32 + n] = (n < 12) ? (_Float16)v : (_Float16)0.0f;
            } else if (MODE == 2) {
                o0[m * D_INNER + n] = v + f0[n];                 // dt_raw + b_dt
            } else if (MODE == 3) {
                int w = (int)(m / WLEN), t = (int)(m % WLEN);
                int bb = w >> 6, win = w & 63;
                long s = (long)bb * 3136 + ((win >> 3) * 7 + t / 7) * 56
                         + ((win & 7) * 7 + t % 7);
                long idx = s * DIMC + n;
                o0[idx] = f0[idx] + v;                           // shortcut + m
            } else if (MODE == 4) {
                float xv = v + f0[n];
                float g  = 0.5f * xv * (1.0f + erff(xv * 0.70710678118f));
                o1[m * 768 + n] = (_Float16)g;
            } else { // MODE 5
                long idx = m * DIMC + n;
                o0[idx] = f1[idx] + v + f0[n];                   // x2 + mlp
            }
        }
    }
}

// ---------------------------------------------------------------------------
// Depthwise causal conv (K=4) + silu, forward and reversed sequence.
// One thread per (window, channel); rolling 3-deep history in registers.
// ---------------------------------------------------------------------------
__global__ __launch_bounds__(256)
void conv_silu_kernel(const float* __restrict__ u,
                      const float* __restrict__ wf, const float* __restrict__ bf,
                      const float* __restrict__ wb, const float* __restrict__ bb,
                      float* __restrict__ ucf, _Float16* __restrict__ ucf16,
                      float* __restrict__ ucb, _Float16* __restrict__ ucb16) {
    long gid = (long)blockIdx.x * 256 + threadIdx.x;
    int w = (int)(gid / D_INNER), d = (int)(gid % D_INNER);
    float w0f = wf[d * 4 + 0], w1f = wf[d * 4 + 1], w2f = wf[d * 4 + 2], w3f = wf[d * 4 + 3];
    float w0b = wb[d * 4 + 0], w1b = wb[d * 4 + 1], w2b = wb[d * 4 + 2], w3b = wb[d * 4 + 3];
    float bfv = bf[d], bbv = bb[d];
    const float* up = u + (long)w * WLEN * D_INNER + d;
    float h0 = 0.f, h1 = 0.f, h2 = 0.f;
    for (int t = 0; t < WLEN; ++t) {
        float cur = up[(long)t * D_INNER];
        float y = w0f * h0 + w1f * h1 + w2f * h2 + w3f * cur + bfv;
        y = y / (1.0f + __expf(-y));
        long o = ((long)w * WLEN + t) * D_INNER + d;
        ucf[o] = y; ucf16[o] = (_Float16)y;
        h0 = h1; h1 = h2; h2 = cur;
    }
    float g0 = 0.f, g1 = 0.f, g2 = 0.f;
    for (int tr = 0; tr < WLEN; ++tr) {
        float cur = up[(long)(48 - tr) * D_INNER];
        float y = w0b * g0 + w1b * g1 + w2b * g2 + w3b * cur + bbv;
        y = y / (1.0f + __expf(-y));
        long o = ((long)w * WLEN + tr) * D_INNER + d;
        ucb[o] = y; ucb16[o] = (_Float16)y;
        g0 = g1; g1 = g2; g2 = cur;
    }
}

// ---------------------------------------------------------------------------
// Selective scan. One block per window, 384 threads (one per channel).
// 16-wide state in registers; B/C (32 floats) via double-buffered LDS.
// dir=1 runs on the reversed sequence and scatters output back un-reversed.
// ---------------------------------------------------------------------------
__global__ __launch_bounds__(384)
void scan_kernel(const float* __restrict__ uc, const float* __restrict__ proj,
                 const float* __restrict__ dtraw,
                 const float* __restrict__ A_log, const float* __restrict__ Dv,
                 float* __restrict__ yout, int dir) {
    __shared__ float bc[2][32];
    const int w = blockIdx.x;
    const int d = threadIdx.x;
    float A[D_STATE];
    #pragma unroll
    for (int n = 0; n < D_STATE; ++n) A[n] = -__expf(A_log[d * D_STATE + n]);
    const float Dd = Dv[d];
    float h[D_STATE];
    #pragma unroll
    for (int n = 0; n < D_STATE; ++n) h[n] = 0.f;

    for (int t = 0; t < WLEN; ++t) {
        const long row = (long)w * WLEN + t;
        if (threadIdx.x < 32)
            bc[t & 1][threadIdx.x] = proj[row * 48 + 12 + threadIdx.x];
        float dtv = dtraw[row * D_INNER + d];
        dtv = (dtv > 20.f) ? dtv : log1pf(__expf(dtv));        // softplus
        float ut = uc[row * D_INNER + d];
        __syncthreads();
        const float* B = bc[t & 1];
        const float* C = bc[t & 1] + 16;
        float y = 0.f;
        #pragma unroll
        for (int n = 0; n < D_STATE; ++n) {
            float dA = __expf(dtv * A[n]);
            h[n] = dA * h[n] + dtv * B[n] * ut;
            y += h[n] * C[n];
        }
        const long orow = dir ? ((long)w * WLEN + (48 - t)) : row;
        yout[orow * D_INNER + d] = y + Dd * ut;
    }
}

// (y_f + y_b) * silu(z)  -> f16 operand for the W_out GEMM
__global__ __launch_bounds__(256)
void combine_kernel(const float* __restrict__ yf, const float* __restrict__ yb,
                    const _Float16* __restrict__ sz, _Float16* __restrict__ yc) {
    long i = (long)blockIdx.x * 256 + threadIdx.x;
    float v = (yf[i] + yb[i]) * (float)sz[i];
    yc[i] = (_Float16)v;
}

// ---------------------------------------------------------------------------
extern "C" void kernel_launch(void* const* d_in, const int* in_sizes, int n_in,
                              void* d_out, int out_size, void* d_ws, size_t ws_size,
                              hipStream_t stream) {
    const float* X    = (const float*)d_in[0];
    const float* g1   = (const float*)d_in[1];
    const float* b1   = (const float*)d_in[2];
    const float* gm   = (const float*)d_in[3];
    const float* bm   = (const float*)d_in[4];
    const float* Win  = (const float*)d_in[5];
    const float* cwf  = (const float*)d_in[6];
    const float* cbf  = (const float*)d_in[7];
    const float* Wxpf = (const float*)d_in[8];
    const float* Wdtf = (const float*)d_in[9];
    const float* bdtf = (const float*)d_in[10];
    const float* Alf  = (const float*)d_in[11];
    const float* Df   = (const float*)d_in[12];
    const float* cwb  = (const float*)d_in[13];
    const float* cbb  = (const float*)d_in[14];
    const float* Wxpb = (const float*)d_in[15];
    const float* Wdtb = (const float*)d_in[16];
    const float* bdtb = (const float*)d_in[17];
    const float* Alb  = (const float*)d_in[18];
    const float* Db   = (const float*)d_in[19];
    const float* Wout = (const float*)d_in[20];
    const float* g2   = (const float*)d_in[21];
    const float* b2   = (const float*)d_in[22];
    const float* Wfc1 = (const float*)d_in[23];
    const float* bfc1 = (const float*)d_in[24];
    const float* Wfc2 = (const float*)d_in[25];
    const float* bfc2 = (const float*)d_in[26];

    // --- workspace carve-up (256B aligned) ---
    char* base = (char*)d_ws;
    size_t off = 0;
    auto take = [&](size_t bytes) -> void* {
        void* p = base + off;
        off += (bytes + 255) & ~(size_t)255;
        return p;
    };
    const long M = NTOK;
    _Float16* hs16   = (_Float16*)take(M * DIMC * 2);
    float*    u      = (float*)   take(M * D_INNER * 4);
    _Float16* sz16   = (_Float16*)take(M * D_INNER * 2);
    float*    ucf    = (float*)   take(M * D_INNER * 4);
    _Float16* ucf16  = (_Float16*)take(M * D_INNER * 2);
    float*    ucb    = (float*)   take(M * D_INNER * 4);
    _Float16* ucb16  = (_Float16*)take(M * D_INNER * 2);
    float*    projf  = (float*)   take(M * 48 * 4);
    float*    projb  = (float*)   take(M * 48 * 4);
    _Float16* dtAf   = (_Float16*)take(M * 32 * 2);
    _Float16* dtAb   = (_Float16*)take(M * 32 * 2);
    float*    dtf    = (float*)   take(M * D_INNER * 4);
    float*    dtb    = (float*)   take(M * D_INNER * 4);
    float*    yfv    = (float*)   take(M * D_INNER * 4);
    float*    ybv    = (float*)   take(M * D_INNER * 4);
    _Float16* yc16   = (_Float16*)take(M * D_INNER * 2);
    float*    x2     = (float*)   take(M * DIMC * 4);
    _Float16* h2_16  = (_Float16*)take(M * DIMC * 2);
    _Float16* act16  = (_Float16*)take(M * 768 * 2);
    _Float16* WtIn   = (_Float16*)take(768 * 192 * 2);
    _Float16* WtXpf  = (_Float16*)take(48 * 384 * 2);
    _Float16* WtXpb  = (_Float16*)take(48 * 384 * 2);
    _Float16* WtDtf  = (_Float16*)take(384 * 32 * 2);
    _Float16* WtDtb  = (_Float16*)take(384 * 32 * 2);
    _Float16* WtOut  = (_Float16*)take(192 * 384 * 2);
    _Float16* WtFc1  = (_Float16*)take(768 * 192 * 2);
    _Float16* WtFc2  = (_Float16*)take(192 * 768 * 2);

    auto pack = [&](const float* W, _Float16* Wt, int K, int N, int Kp, int Np) {
        int total = Kp * Np;
        pack_weight<<<(total + 255) / 256, 256, 0, stream>>>(W, Wt, K, N, Kp, Np);
    };
    pack(Win,  WtIn,  192, 768, 192, 768);
    pack(Wxpf, WtXpf, 384,  44, 384,  48);
    pack(Wxpb, WtXpb, 384,  44, 384,  48);
    pack(Wdtf, WtDtf,  12, 384,  32, 384);
    pack(Wdtb, WtDtb,  12, 384,  32, 384);
    pack(Wout, WtOut, 384, 192, 384, 192);
    pack(Wfc1, WtFc1, 192, 768, 192, 768);
    pack(Wfc2, WtFc2, 768, 192, 768, 192);

    // LN1 -> window partition -> LN2 -> f16
    ln_window_kernel<<<(int)M, 192, 0, stream>>>(X, g1, b1, gm, bm, hs16);

    // xz = hs @ W_in : split into u (fp32) / silu(z) (f16).  48 N-tiles = 12 strips
    wmma_gemm<0, 4, 192><<<MTILES * 12 / 8, 256, 0, stream>>>(hs16, WtIn, 12,
                                                              nullptr, nullptr, u, sz16);
    // depthwise causal conv + silu (fwd + reversed)
    conv_silu_kernel<<<NWIN * D_INNER / 256, 256, 0, stream>>>(
        u, cwf, cbf, cwb, cbb, ucf, ucf16, ucb, ucb16);

    // proj = uc @ W_xp  (N padded 44 -> 48 = one 3-tile strip), + f16 dt-A operand
    wmma_gemm<1, 3, 384><<<MTILES / 8, 256, 0, stream>>>(ucf16, WtXpf, 1,
                                                         nullptr, nullptr, projf, dtAf);
    wmma_gemm<1, 3, 384><<<MTILES / 8, 256, 0, stream>>>(ucb16, WtXpb, 1,
                                                         nullptr, nullptr, projb, dtAb);

    // dt_raw = proj[:, :12] @ W_dt + b_dt   (K padded 12 -> 32; 24 tiles = 6 strips)
    wmma_gemm<2, 4, 32><<<MTILES * 6 / 8, 256, 0, stream>>>(dtAf, WtDtf, 6,
                                                            bdtf, nullptr, dtf, nullptr);
    wmma_gemm<2, 4, 32><<<MTILES * 6 / 8, 256, 0, stream>>>(dtAb, WtDtb, 6,
                                                            bdtb, nullptr, dtb, nullptr);

    // selective scans
    scan_kernel<<<NWIN, 384, 0, stream>>>(ucf, projf, dtf, Alf, Df, yfv, 0);
    scan_kernel<<<NWIN, 384, 0, stream>>>(ucb, projb, dtb, Alb, Db, ybv, 1);

    // y = (y_f + y_b) * silu(z)
    combine_kernel<<<(int)(M * D_INNER / 256), 256, 0, stream>>>(yfv, ybv, sz16, yc16);

    // x2 = shortcut + window_reverse(y @ W_out)   (12 tiles = 3 strips)
    wmma_gemm<3, 4, 384><<<MTILES * 3 / 8, 256, 0, stream>>>(yc16, WtOut, 3,
                                                             X, nullptr, x2, nullptr);
    // LN2
    ln_kernel<<<(int)M, 192, 0, stream>>>(x2, g2, b2, h2_16);

    // act = gelu(h2 @ W_fc1 + b1) -> f16   (48 tiles = 12 strips)
    wmma_gemm<4, 4, 192><<<MTILES * 12 / 8, 256, 0, stream>>>(h2_16, WtFc1, 12,
                                                              bfc1, nullptr, nullptr, act16);
    // out = x2 + act @ W_fc2 + b2   (12 tiles = 3 strips)
    wmma_gemm<5, 4, 768><<<MTILES * 3 / 8, 256, 0, stream>>>(act16, WtFc2, 3,
                                                             bfc2, x2, (float*)d_out, nullptr);
}
